// S4Layer_17927193493900
// MI455X (gfx1250) — matrix operand order repaired and checked by reading, not compiled, not run on hardware
//
#include <hip/hip_runtime.h>
#include <hip/hip_bf16.h>
#include <stdint.h>

#define L_SEQ   2048
#define DMODEL  1024
#define DINNER  2048
#define DSTATE  16
#define DCONV   4
#define DTRANK  64

#define USE_ASYNC_COPY 1

typedef __attribute__((ext_vector_type(16))) __bf16          v16bf;
typedef __attribute__((ext_vector_type(16))) unsigned short  v16u;
typedef __attribute__((ext_vector_type(8)))  float           v8f;

__device__ __forceinline__ unsigned short f32_to_bf16_rne(float f) {
    unsigned int x = __float_as_uint(f);
    unsigned int r = x + 0x7FFFu + ((x >> 16) & 1u);
    return (unsigned short)(r >> 16);
}
__device__ __forceinline__ float bf16_to_f32(unsigned short u) {
    return __uint_as_float(((unsigned int)u) << 16);
}

// 16-byte global -> LDS copy. Async variant uses the CDNA5 ASYNCcnt path
// (per-lane contiguous b128, LDS address in VDST VGPR per ISA 15.18.3).
__device__ __forceinline__ void copy16_g2l(unsigned short* lds_dst,
                                           const unsigned short* gsrc) {
#if USE_ASYNC_COPY
    unsigned int loff = (unsigned int)(uintptr_t)lds_dst;   // low 32 bits = LDS offset
    asm volatile("global_load_async_to_lds_b128 %0, %1, off"
                 :: "v"(loff), "v"(gsrc) : "memory");
#else
    *(uint4*)lds_dst = *(const uint4*)gsrc;
#endif
}
__device__ __forceinline__ void async_drain() {
#if USE_ASYNC_COPY
    asm volatile("s_wait_asynccnt 0" ::: "memory");
#endif
}

// ---------------------------------------------------------------------------
// Decompose f32 array into bf16 hi/lo pair (for bf16x3 WMMA GEMMs)
// ---------------------------------------------------------------------------
__global__ void decompose_kernel(const float* __restrict__ src, int n,
                                 unsigned short* __restrict__ hi,
                                 unsigned short* __restrict__ lo) {
    int i = blockIdx.x * blockDim.x + threadIdx.x;
    if (i >= n) return;
    float v = src[i];
    unsigned short h = f32_to_bf16_rne(v);
    hi[i] = h;
    lo[i] = f32_to_bf16_rne(v - bf16_to_f32(h));
}

// ---------------------------------------------------------------------------
// Tiled GEMM: C[M,N] = A[M,K]*B[K,N] + bias, f32 accuracy via bf16x3 WMMA.
// Block = 256 thr (8 waves, 2x4), block tile 128x128, wave tile 64x32,
// K-step 32, double-buffered async global->LDS staging.
// A is staged in WMMA-fragment order (16B/lane chunks); B row-major + pad.
// Requires M%128==0, N%128==0, K%32==0.
// ---------------------------------------------------------------------------
__global__ __launch_bounds__(256) void gemm_bf16x3_tiled(
        const unsigned short* __restrict__ Ah, const unsigned short* __restrict__ Al,
        const unsigned short* __restrict__ Bh, const unsigned short* __restrict__ Bl,
        const float* __restrict__ bias, float* __restrict__ C,
        int M, int N, int K) {
    __shared__ __align__(32) unsigned short sA[2][2][8 * 32 * 16]; // [buf][hi/lo][frag order]
    __shared__ __align__(32) unsigned short sB[2][2][32 * 136];    // [buf][hi/lo][k][n+pad]

    const int tid    = threadIdx.x;
    const int lane   = tid & 31;
    const int wave   = tid >> 5;
    const int waveM  = wave >> 2;        // 0..1
    const int waveN  = wave & 3;         // 0..3
    const int laneLo = lane & 15;
    const int laneHi = lane >> 4;
    const int bm     = blockIdx.y * 128;
    const int bn     = blockIdx.x * 128;

    v8f acc[4][2] = {};

    // ---- staging: fill buffer `buf` with K-slice starting at k0 ----
    auto stage = [&](int buf, int k0) {
        // A tile 128x32, fragment order: lds[(sub*32+lane)*16 + g*8 .. +7]
        //   octet o: row=o>>2, g=(o>>1)&1, h=o&1  (g->j-half, h->laneHi)
#pragma unroll
        for (int arr = 0; arr < 2; ++arr) {
            const unsigned short* src = arr ? Al : Ah;
            unsigned short* dst = sA[buf][arr];
#pragma unroll
            for (int it = 0; it < 2; ++it) {
                int o   = tid + 256 * it;
                int row = o >> 2, g = (o >> 1) & 1, h = o & 1;
                const unsigned short* gp =
                    src + (size_t)(bm + row) * K + k0 + g * 16 + h * 8;
                unsigned short* lp =
                    dst + (((row >> 4) * 32 + (row & 15) + 16 * h) * 16 + g * 8);
                copy16_g2l(lp, gp);
            }
        }
        // B tile 32x128 row-major (pitch 136)
#pragma unroll
        for (int arr = 0; arr < 2; ++arr) {
            const unsigned short* src = arr ? Bl : Bh;
            unsigned short* dst = sB[buf][arr];
#pragma unroll
            for (int it = 0; it < 2; ++it) {
                int e   = tid + 256 * it;
                int row = e >> 4, seg = e & 15;
                const unsigned short* gp = src + (size_t)(k0 + row) * N + bn + seg * 8;
                unsigned short* lp = dst + row * 136 + seg * 8;
                copy16_g2l(lp, gp);
            }
        }
    };

    stage(0, 0);
    async_drain();
    __syncthreads();

    const int nk = K >> 5;
    for (int ki = 0; ki < nk; ++ki) {
        const int cur = ki & 1;
        if (ki + 1 < nk) stage(cur ^ 1, (ki + 1) << 5);   // overlap next-tile copy

        // ---- B fragments for this wave's two n-subtiles ----
        v16u bhu[2], blu[2];
#pragma unroll
        for (int t = 0; t < 2; ++t) {
            const int col = (waveN * 2 + t) * 16 + laneLo;
#pragma unroll
            for (int j = 0; j < 16; ++j) {
                const int row = ((j >= 8) ? 16 : 0) + laneHi * 8 + (j & 7);           // k within 32-slice
                bhu[t][j] = sB[cur][0][row * 136 + col];
                blu[t][j] = sB[cur][1][row * 136 + col];
            }
        }
        // ---- 4 m-subtiles x 2 n-subtiles x 3 WMMA ----
#pragma unroll
        for (int s = 0; s < 4; ++s) {
            const int sub = waveM * 4 + s;
            v16u ahu = *(const v16u*)&sA[cur][0][(sub * 32 + lane) * 16];
            v16u alu = *(const v16u*)&sA[cur][1][(sub * 32 + lane) * 16];
            v16bf ah = __builtin_bit_cast(v16bf, ahu);
            v16bf al = __builtin_bit_cast(v16bf, alu);
#pragma unroll
            for (int t = 0; t < 2; ++t) {
                v16bf bh = __builtin_bit_cast(v16bf, bhu[t]);
                v16bf bl = __builtin_bit_cast(v16bf, blu[t]);
                acc[s][t] = __builtin_amdgcn_wmma_f32_16x16x32_bf16(false, ah, false, bh, (short)0, acc[s][t], false, false);
                acc[s][t] = __builtin_amdgcn_wmma_f32_16x16x32_bf16(false, ah, false, bl, (short)0, acc[s][t], false, false);
                acc[s][t] = __builtin_amdgcn_wmma_f32_16x16x32_bf16(false, al, false, bh, (short)0, acc[s][t], false, false);
            }
        }
        async_drain();
        __syncthreads();
    }

    // ---- write back: C/D layout -> row = sub*16 + laneHi*8 + i, col = laneLo ----
#pragma unroll
    for (int s = 0; s < 4; ++s) {
#pragma unroll
        for (int t = 0; t < 2; ++t) {
            const int col = bn + (waveN * 2 + t) * 16 + laneLo;
            const float b = bias[col];
#pragma unroll
            for (int i = 0; i < 8; ++i) {
                const int row = bm + (waveM * 4 + s) * 16 + laneHi * 8 + i;
                C[(size_t)row * N + col] = acc[s][t][i] + b;
            }
        }
    }
}

// ---------------------------------------------------------------------------
// Simple per-wave 16x16 GEMM (bf16x3 WMMA) for the skinny dt projections.
// ---------------------------------------------------------------------------
__global__ void gemm_bf16x3_small(const unsigned short* __restrict__ Ah,
                                  const unsigned short* __restrict__ Al,
                                  const unsigned short* __restrict__ Bh,
                                  const unsigned short* __restrict__ Bl,
                                  const float* __restrict__ bias,
                                  float* __restrict__ C,
                                  int M, int N, int K) {
    const int lane   = threadIdx.x & 31;
    const int wave   = threadIdx.x >> 5;
    const int tilesN = N >> 4;
    const int tile   = blockIdx.x * 8 + wave;
    if (tile >= (M >> 4) * tilesN) return;
    const int tm = tile / tilesN;
    const int tn = tile % tilesN;
    const int laneLo = lane & 15;
    const int laneHi = lane >> 4;
    const int m = tm * 16 + laneLo;
    const int n = tn * 16 + laneLo;

    v8f acc = {};
    for (int k0 = 0; k0 < K; k0 += 32) {
        v16u ahu, alu, bhu, blu;
#pragma unroll
        for (int j = 0; j < 16; ++j) {
            int ka = k0 + ((j >= 8) ? 16 : 0) + laneHi * 8 + (((j >> 1) & 3) << 1) + (j & 1);
            int kb = k0 + ((j >= 8) ? 16 : 0) + laneHi * 8 + (j & 7);
            ahu[j] = Ah[m * K + ka];
            alu[j] = Al[m * K + ka];
            bhu[j] = Bh[kb * N + n];
            blu[j] = Bl[kb * N + n];
        }
        v16bf ah = __builtin_bit_cast(v16bf, ahu);
        v16bf al = __builtin_bit_cast(v16bf, alu);
        v16bf bh = __builtin_bit_cast(v16bf, bhu);
        v16bf bl = __builtin_bit_cast(v16bf, blu);
        acc = __builtin_amdgcn_wmma_f32_16x16x32_bf16(false, ah, false, bh, (short)0, acc, false, false);
        acc = __builtin_amdgcn_wmma_f32_16x16x32_bf16(false, ah, false, bl, (short)0, acc, false, false);
        acc = __builtin_amdgcn_wmma_f32_16x16x32_bf16(false, al, false, bh, (short)0, acc, false, false);
    }
    const float b = bias[n];
#pragma unroll
    for (int i = 0; i < 8; ++i) {
        int row = tm * 16 + laneHi * 8 + i;
        C[row * N + n] = acc[i] + b;
    }
}

// ---------------------------------------------------------------------------
// Causal depthwise conv1d (width 4) + SiLU; also emits bf16 hi/lo of result
// ---------------------------------------------------------------------------
__global__ void conv_silu_kernel(const float* __restrict__ u_pre,
                                 const float* __restrict__ conv_w,
                                 const float* __restrict__ conv_b,
                                 float* __restrict__ u,
                                 unsigned short* __restrict__ u_hi,
                                 unsigned short* __restrict__ u_lo) {
    int idx = blockIdx.x * blockDim.x + threadIdx.x;
    if (idx >= L_SEQ * DINNER) return;
    int l = idx / DINNER;
    int d = idx - l * DINNER;
    float acc = conv_b[d];
#pragma unroll
    for (int j = 0; j < DCONV; ++j) {
        int ll = l - (DCONV - 1) + j;
        if (ll >= 0) acc = fmaf(u_pre[ll * DINNER + d], conv_w[d * DCONV + j], acc);
    }
    float s = acc / (1.0f + expf(-acc));   // SiLU
    u[idx] = s;
    unsigned short h = f32_to_bf16_rne(s);
    u_hi[idx] = h;
    u_lo[idx] = f32_to_bf16_rne(s - bf16_to_f32(h));
}

// ---------------------------------------------------------------------------
// Fused selective-scan: softplus -> ZOH -> recurrence -> C contraction -> +D*u
// Lane = state n within 16-lane groups (wave32: 2 channels/wave).
// ---------------------------------------------------------------------------
__global__ void scan_kernel(const float* __restrict__ delta_pre,
                            const float* __restrict__ u,
                            const float* __restrict__ A_log,
                            const float* __restrict__ B_mat,
                            const float* __restrict__ C_mat,
                            const float* __restrict__ D_vec,
                            float* __restrict__ y_ssm) {
    int gid = blockIdx.x * blockDim.x + threadIdx.x;
    int d = gid >> 4;
    int n = gid & 15;
    if (d >= DINNER) return;

    const float A    = -expf(A_log[n]);
    const float Bn   = B_mat[n * DINNER + d];
    const float Cn   = C_mat[n * DINNER + d];
    const float Dd   = D_vec[d];
    const bool  tiny = fabsf(A) < 1e-6f;
    const float invA = tiny ? 0.0f : (1.0f / A);

    float h = 0.0f;
    for (int l = 0; l < L_SEQ; ++l) {
        if (l + 8 < L_SEQ) {                       // hide HBM latency of serial chain
            __builtin_prefetch(&delta_pre[(l + 8) * DINNER + d], 0, 0);
            __builtin_prefetch(&u[(l + 8) * DINNER + d], 0, 0);
        }
        float dp = delta_pre[l * DINNER + d];
        float uu = u[l * DINNER + d];
        float dlt = (dp > 20.0f) ? dp : log1pf(expf(dp));   // softplus
        float dA = dlt * A;
        float ab = expf(dA);                                 // A_bar
        float ex = tiny ? dlt * (1.0f + 0.5f * dA) : (ab - 1.0f) * invA;
        h = fmaf(ab, h, ex * Bn * uu);
        float p = h * Cn;
        p += __shfl_xor(p, 1);
        p += __shfl_xor(p, 2);
        p += __shfl_xor(p, 4);
        p += __shfl_xor(p, 8);
        if (n == 0) y_ssm[l * DINNER + d] = p + Dd * uu;
    }
}

// ---------------------------------------------------------------------------
// Gating: y = y_ssm * silu(g_pre); emit bf16 hi/lo for the output GEMM
// ---------------------------------------------------------------------------
__global__ void gate_kernel(const float* __restrict__ y_ssm,
                            const float* __restrict__ g_pre,
                            unsigned short* __restrict__ yg_hi,
                            unsigned short* __restrict__ yg_lo) {
    int idx = blockIdx.x * blockDim.x + threadIdx.x;
    if (idx >= L_SEQ * DINNER) return;
    float g = g_pre[idx];
    float y = y_ssm[idx] * (g / (1.0f + expf(-g)));
    unsigned short h = f32_to_bf16_rne(y);
    yg_hi[idx] = h;
    yg_lo[idx] = f32_to_bf16_rne(y - bf16_to_f32(h));
}

// ---------------------------------------------------------------------------
extern "C" void kernel_launch(void* const* d_in, const int* in_sizes, int n_in,
                              void* d_out, int out_size, void* d_ws, size_t ws_size,
                              hipStream_t stream) {
    const float* x      = (const float*)d_in[0];
    const float* in_w   = (const float*)d_in[1];
    const float* in_b   = (const float*)d_in[2];
    const float* conv_w = (const float*)d_in[3];
    const float* conv_b = (const float*)d_in[4];
    const float* A_log  = (const float*)d_in[5];
    const float* B_mat  = (const float*)d_in[6];
    const float* C_mat  = (const float*)d_in[7];
    const float* D_vec  = (const float*)d_in[8];
    const float* gate_w = (const float*)d_in[9];
    const float* gate_b = (const float*)d_in[10];
    const float* dt1_w  = (const float*)d_in[11];
    const float* dt1_b  = (const float*)d_in[12];
    const float* dt2_w  = (const float*)d_in[13];
    const float* dt2_b  = (const float*)d_in[14];
    const float* out_w  = (const float*)d_in[15];
    const float* out_b  = (const float*)d_in[16];
    float* out = (float*)d_out;

    const size_t LD  = (size_t)L_SEQ * DINNER;
    const size_t LDM = (size_t)L_SEQ * DMODEL;
    const size_t WIN = (size_t)DMODEL * DINNER;
    const size_t WOU = (size_t)DINNER * DMODEL;
    const size_t WD1 = (size_t)DINNER * DTRANK;
    const size_t WD2 = (size_t)DTRANK * DINNER;
    const size_t LR  = (size_t)L_SEQ * DTRANK;

    uint8_t* p = (uint8_t*)d_ws;
    auto alloc = [&](size_t bytes) -> uint8_t* {
        uint8_t* r = p;
        p += (bytes + 255) & ~(size_t)255;
        return r;
    };
    float* u_pre     = (float*)alloc(LD * 4);     // reused as y_ssm after conv
    float* g_pre     = (float*)alloc(LD * 4);
    float* u_buf     = (float*)alloc(LD * 4);
    float* delta_pre = (float*)alloc(LD * 4);
    float* dtlow     = (float*)alloc(LR * 4);
    unsigned short* x_hi  = (unsigned short*)alloc(LDM * 2);
    unsigned short* x_lo  = (unsigned short*)alloc(LDM * 2);
    unsigned short* inw_h = (unsigned short*)alloc(WIN * 2);
    unsigned short* inw_l = (unsigned short*)alloc(WIN * 2);
    unsigned short* gw_h  = (unsigned short*)alloc(WIN * 2);
    unsigned short* gw_l  = (unsigned short*)alloc(WIN * 2);
    unsigned short* ow_h  = (unsigned short*)alloc(WOU * 2);
    unsigned short* ow_l  = (unsigned short*)alloc(WOU * 2);
    unsigned short* d1_h  = (unsigned short*)alloc(WD1 * 2);
    unsigned short* d1_l  = (unsigned short*)alloc(WD1 * 2);
    unsigned short* d2_h  = (unsigned short*)alloc(WD2 * 2);
    unsigned short* d2_l  = (unsigned short*)alloc(WD2 * 2);
    unsigned short* u_hi  = (unsigned short*)alloc(LD * 2);
    unsigned short* u_lo  = (unsigned short*)alloc(LD * 2);
    unsigned short* dtl_h = (unsigned short*)alloc(LR * 2);
    unsigned short* dtl_l = (unsigned short*)alloc(LR * 2);
    unsigned short* yg_hi = (unsigned short*)alloc(LD * 2);
    unsigned short* yg_lo = (unsigned short*)alloc(LD * 2);
    float* y_ssm = u_pre;

    auto dgrid = [](size_t n) { return (unsigned)((n + 255) / 256); };
    auto sgrid = [](int M, int N) { return (unsigned)(((size_t)(M / 16) * (N / 16) + 7) / 8); };

    // 1) bf16 hi/lo decompositions
    decompose_kernel<<<dgrid(LDM), 256, 0, stream>>>(x,      (int)LDM, x_hi, x_lo);
    decompose_kernel<<<dgrid(WIN), 256, 0, stream>>>(in_w,   (int)WIN, inw_h, inw_l);
    decompose_kernel<<<dgrid(WIN), 256, 0, stream>>>(gate_w, (int)WIN, gw_h, gw_l);
    decompose_kernel<<<dgrid(WOU), 256, 0, stream>>>(out_w,  (int)WOU, ow_h, ow_l);
    decompose_kernel<<<dgrid(WD1), 256, 0, stream>>>(dt1_w,  (int)WD1, d1_h, d1_l);
    decompose_kernel<<<dgrid(WD2), 256, 0, stream>>>(dt2_w,  (int)WD2, d2_h, d2_l);

    // 2) u_pre = x @ in_w + in_b ; g_pre = x @ gate_w + gate_b   (tiled WMMA)
    gemm_bf16x3_tiled<<<dim3(DINNER / 128, L_SEQ / 128), 256, 0, stream>>>(
        x_hi, x_lo, inw_h, inw_l, in_b, u_pre, L_SEQ, DINNER, DMODEL);
    gemm_bf16x3_tiled<<<dim3(DINNER / 128, L_SEQ / 128), 256, 0, stream>>>(
        x_hi, x_lo, gw_h, gw_l, gate_b, g_pre, L_SEQ, DINNER, DMODEL);

    // 3) causal depthwise conv + SiLU (+ bf16 split of u)
    conv_silu_kernel<<<dgrid(LD), 256, 0, stream>>>(u_pre, conv_w, conv_b, u_buf, u_hi, u_lo);

    // 4) delta_pre = (u @ dt1_w + dt1_b) @ dt2_w + dt2_b   (skinny: simple WMMA)
    gemm_bf16x3_small<<<sgrid(L_SEQ, DTRANK), 256, 0, stream>>>(
        u_hi, u_lo, d1_h, d1_l, dt1_b, dtlow, L_SEQ, DTRANK, DINNER);
    decompose_kernel<<<dgrid(LR), 256, 0, stream>>>(dtlow, (int)LR, dtl_h, dtl_l);
    gemm_bf16x3_small<<<sgrid(L_SEQ, DINNER), 256, 0, stream>>>(
        dtl_h, dtl_l, d2_h, d2_l, dt2_b, delta_pre, L_SEQ, DINNER, DTRANK);

    // 5) fused selective scan
    scan_kernel<<<(DINNER * DSTATE) / 256, 256, 0, stream>>>(
        delta_pre, u_buf, A_log, B_mat, C_mat, D_vec, y_ssm);

    // 6) gate with silu(g_pre)
    gate_kernel<<<dgrid(LD), 256, 0, stream>>>(y_ssm, g_pre, yg_hi, yg_lo);

    // 7) out = y_gated @ out_w + out_b   (tiled WMMA, writes all of d_out)
    gemm_bf16x3_tiled<<<dim3(DMODEL / 128, L_SEQ / 128), 256, 0, stream>>>(
        yg_hi, yg_lo, ow_h, ow_l, out_b, out, L_SEQ, DMODEL, DINNER);
}
